// MultiHeadAttention_3573412790522
// MI455X (gfx1250) — matrix operand order repaired and checked by reading, not compiled, not run on hardware
//
#include <hip/hip_runtime.h>

// ---------------------------------------------------------------------------
// MultiHeadAttention forward for MI455X (gfx1250, wave32, WMMA).
//  - All GEMMs: v_wmma_f32_16x16x32_bf16 (bf16 in, fp32 accumulate).
//  - QK^T + mask + softmax + A@V fused: the 16x2048 fp32 score strip lives in
//    LDS, so the attention matrix touches HBM exactly once (the mandatory
//    1.07 GB output write) instead of 4x (4.3 GB) unfused.
//  - K-tiles for QK^T are double-buffered into LDS with
//    GLOBAL_LOAD_ASYNC_TO_LDS_B128 (ASYNCcnt), prefetching tile jt+8 while
//    the WMMAs consume tile jt.
// ---------------------------------------------------------------------------

typedef __bf16 bf16;
typedef __attribute__((ext_vector_type(16))) __bf16 v16bf;
typedef __attribute__((ext_vector_type(8)))  __bf16 v8bf;
typedef __attribute__((ext_vector_type(8)))  float  v8f;
typedef __attribute__((ext_vector_type(4)))  float  v4f;

static constexpr int B  = 4;
static constexpr int S  = 2048;
static constexpr int D  = 1024;
static constexpr int H  = 16;
static constexpr int HD = 64;

// LDS strip row stride (floats): 2048 + 4 so stride % 64 banks == 4 ->
// the 16 concurrently-read A-fragment rows map to distinct bank groups.
static constexpr int LST = S + 4;                 // 2052
// strip (16*LST) + k-half scratch (1024) + 8 waves * (2 x 2KB) K-tile stage
static constexpr int SMEM_FLOATS = 16 * LST + 1024 + 8 * 1024;   // 168,192 B

// ---------------------------------------------------------------------------
// Fragment loaders (contiguous k order, shared by A and B).
// A-fragment (16x32 bf16): lane = (row%16) + 16*khalf.
//   elems 0..15 -> K = kbase + 16*khalf + (0..15)
// B-fragment (32x16 bf16): lane = (col%16) + 16*kgroup.
//   elems 0..15 -> K = kbase + 16*kgroup + (0..15)
// All B matrices are stored [N][K] (W rows / K rows / V^T rows) -> one
// contiguous 32-byte row segment per lane, no transpose needed.
// ---------------------------------------------------------------------------

__device__ __forceinline__ v16bf load_a_bf16(const bf16* __restrict__ base, int lda,
                                             int row, int kbase, int lane) {
    const int khalf = lane >> 4;
    const bf16* p = base + (size_t)row * lda + kbase + 16 * khalf;
    v8bf lo = *(const v8bf*)p;
    v8bf hi = *(const v8bf*)(p + 8);
    return __builtin_shufflevector(lo, hi, 0,1,2,3,4,5,6,7,8,9,10,11,12,13,14,15);
}

// A-fragment from an fp32 LDS strip (16-byte aligned chunks), cvt to bf16.
__device__ __forceinline__ v16bf load_a_lds_f32(const float* strip, int row,
                                                int kbase, int lane) {
    const int khalf = lane >> 4;
    const float* p = strip + row * LST + kbase + 16 * khalf;
    v4f x0 = *(const v4f*)p;
    v4f x1 = *(const v4f*)(p + 4);
    v4f x2 = *(const v4f*)(p + 8);
    v4f x3 = *(const v4f*)(p + 12);
    v8f lo = __builtin_shufflevector(x0, x1, 0,1,2,3,4,5,6,7);
    v8f hi = __builtin_shufflevector(x2, x3, 0,1,2,3,4,5,6,7);
    v8bf lob = __builtin_convertvector(lo, v8bf);
    v8bf hib = __builtin_convertvector(hi, v8bf);
    return __builtin_shufflevector(lob, hib, 0,1,2,3,4,5,6,7,8,9,10,11,12,13,14,15);
}

__device__ __forceinline__ v16bf load_b_rows(const bf16* __restrict__ base, int ldb,
                                             int col, int kbase, int lane) {
    const int kgroup = lane >> 4;
    const bf16* p = base + (size_t)col * ldb + kbase + 16 * kgroup;
    return *(const v16bf*)p;
}

__device__ __forceinline__ v8f wmma_bf16(v16bf a, v16bf b, v8f c) {
    return __builtin_amdgcn_wmma_f32_16x16x32_bf16(false, a, false, b,
                                                   (short)0, c, false, false);
}

// Async copy of one contiguous 2KB block (global -> LDS), 16B per lane, 4 ops.
// INST_OFFSET is added to both the LDS and the global address (ISA 10.2/§4.4).
__device__ __forceinline__ void async_copy_2k(unsigned ldsOff, const char* gsrc,
                                              int lane) {
    unsigned l = ldsOff + lane * 16;
    const char* g = gsrc + lane * 16;
    asm volatile(
        "global_load_async_to_lds_b128 %0, %1, off\n\t"
        "global_load_async_to_lds_b128 %0, %1, off offset:512\n\t"
        "global_load_async_to_lds_b128 %0, %1, off offset:1024\n\t"
        "global_load_async_to_lds_b128 %0, %1, off offset:1536"
        :: "v"(l), "v"(g) : "memory");
}

// ---------------------------------------------------------------------------
// fp32 -> bf16 elementwise conversion.
// ---------------------------------------------------------------------------
__global__ __launch_bounds__(256) void cvt_f32_bf16(const float* __restrict__ in,
                                                    bf16* __restrict__ out, int n) {
    int i = blockIdx.x * 256 + threadIdx.x;
    if (i < n) out[i] = (bf16)in[i];
}

// ---------------------------------------------------------------------------
// Generic GEMM: out[m,n] = sum_k A[m,k] * W[n,k] + bias[n]
// One 16x64 tile per wave (4 accumulators share each A-fragment -> 4 back-to-
// back independent WMMAs per K-step). 8 waves / 256 threads per block.
//   MODE 0: out = bf16, heads layout  [b, h, s, hd]   (Q, K projections)
//   MODE 1: out = bf16, transposed    [b, h, hd, s]   (V projection)
//   MODE 2: out = fp32, flat          [m, n]          (final O projection)
// ---------------------------------------------------------------------------
template <int MODE>
__global__ __launch_bounds__(256) void gemm_xwT(const bf16* __restrict__ A,
                                                const bf16* __restrict__ W,
                                                const float* __restrict__ bias,
                                                void* __restrict__ outp,
                                                int M, int N, int K) {
    const int lane = threadIdx.x & 31;
    const int wave = threadIdx.x >> 5;
    const int tilesN4 = N >> 6;                 // 16x64 tiles along N
    const int tile = blockIdx.x * 8 + wave;
    if (tile >= (M >> 4) * tilesN4) return;     // wave-uniform: EXEC stays full
    const int mt  = tile / tilesN4;
    const int nt4 = tile - mt * tilesN4;

    const int colbase = lane & 15;
    const int arow = mt * 16 + colbase;
    const bf16* aRowPtr = A + (size_t)arow * K;

    v8f acc[4] = {v8f{}, v8f{}, v8f{}, v8f{}};
    for (int k = 0; k < K; k += 32) {
        // pull the next K-chunk of the streaming A row toward the WGP
        __builtin_prefetch(aRowPtr + k + 64, 0, 1);
        v16bf a = load_a_bf16(A, K, arow, k, lane);
#pragma unroll
        for (int t = 0; t < 4; ++t) {
            v16bf b = load_b_rows(W, K, nt4 * 64 + t * 16 + colbase, k, lane);
            acc[t] = wmma_bf16(a, b, acc[t]);
        }
    }

    const int mbase = mt * 16 + ((lane >> 4) << 3);  // C layout: vgpr r -> M=8*(lane/16)+r
#pragma unroll
    for (int t = 0; t < 4; ++t) {
        const int n  = nt4 * 64 + t * 16 + colbase;
        const float bv = bias[n];
#pragma unroll
        for (int r = 0; r < 8; ++r) {
            const int m = mbase + r;
            const float val = acc[t][r] + bv;
            if (MODE == 2) {
                ((float*)outp)[(size_t)m * N + n] = val;
            } else {
                const int bb = m >> 11;          // m / S
                const int s  = m & (S - 1);
                const int h  = n >> 6;           // n / HD
                const int hd = n & (HD - 1);
                size_t idx;
                if (MODE == 0) idx = ((size_t)((bb * H + h) * S + s)) * HD + hd;
                else           idx = ((size_t)((bb * H + h) * HD + hd)) * S + s;
                ((bf16*)outp)[idx] = (bf16)val;
            }
        }
    }
}

// ---------------------------------------------------------------------------
// Fused attention core. One workgroup (8 waves) owns a 16-row query strip:
//   phase 1: energy[16][2048] = (q k^T)/8, masked -> LDS strip (fp32).
//            K-tiles (contiguous 2KB each) double-buffered into LDS with
//            GLOBAL_LOAD_ASYNC_TO_LDS_B128 + s_wait_asynccnt.
//   phase 2: row softmax in LDS; final attention written to d_out once.
//   phase 3: x[16][64] = P @ V, A-fragments straight from LDS.
// Grid: B*H*(S/16) = 8192 workgroups; ~168 KB dynamic LDS each (<320 KB WGP).
// ---------------------------------------------------------------------------
__global__ __launch_bounds__(256) void attn_fused(const bf16* __restrict__ q,
                                                  const bf16* __restrict__ kk,
                                                  const bf16* __restrict__ vT,
                                                  const int*  __restrict__ mask,
                                                  float* __restrict__ attn,
                                                  bf16*  __restrict__ xout) {
    extern __shared__ float smem[];
    float* strip   = smem;                      // [16][LST] fp32 scores
    float* scratch = smem + 16 * LST;           // [4][256] k-half reduction
    char*  kstage  = (char*)(smem + 16 * LST + 1024);   // 8 waves * 2 * 2KB

    const int lane = threadIdx.x & 31;
    const int wave = threadIdx.x >> 5;
    const int bh = blockIdx.x >> 7;             // / (S/16)
    const int it = blockIdx.x & 127;
    const int bb = bh >> 4;                     // batch
    const int h  = bh & 15;                     // head

    const int colbase = lane & 15;
    const int rbase   = (lane >> 4) << 3;

    // ---- Phase 1: energy strip ------------------------------------------
    const bf16* qb = q  + (size_t)bh * S * HD;
    const bf16* kb = kk + (size_t)bh * S * HD;
    const int arow = it * 16 + colbase;
    v16bf a0 = load_a_bf16(qb, HD, arow, 0,  lane);   // reused for all 16 j-tiles
    v16bf a1 = load_a_bf16(qb, HD, arow, 32, lane);

    char* mybuf = kstage + wave * 4096;                  // two 2KB buffers
    const unsigned myOff = (unsigned)(uintptr_t)mybuf;   // LDS byte offset

    // preload buffer 0 with this wave's first K-tile (rows jt*16.., 2KB contig)
    async_copy_2k(myOff, (const char*)kb + (size_t)wave * 2048, lane);

    for (int jj = 0; jj < 16; ++jj) {           // uniform trip count: no divergence
        const int jt  = wave + jj * 8;
        const int buf = jj & 1;
        if (jj < 15) {                          // prefetch next tile into other buffer
            const int jn = wave + (jj + 1) * 8;
            async_copy_2k(myOff + (buf ^ 1) * 2048,
                          (const char*)kb + (size_t)jn * 2048, lane);
            asm volatile("s_wait_asynccnt 0x4" ::: "memory");   // current buf done
        } else {
            asm volatile("s_wait_asynccnt 0x0" ::: "memory");
        }
        // B-fragments from the staged tile: row = colbase, 64 bf16 per row
        const bf16* kt = (const bf16*)(mybuf + buf * 2048);
        const int kg16 = (lane >> 4) * 16;
        v16bf b0 = *(const v16bf*)(kt + colbase * 64 + kg16);
        v16bf b1 = *(const v16bf*)(kt + colbase * 64 + 32 + kg16);
        v8f acc = {};
        acc = wmma_bf16(a0, b0, acc);
        acc = wmma_bf16(a1, b1, acc);
        const int mz = mask[bb * S + jt * 16 + colbase];
#pragma unroll
        for (int r = 0; r < 8; ++r) {
            const float e = (mz == 0) ? -1e10f : acc[r] * 0.125f;  // 1/sqrt(64)
            strip[(rbase + r) * LST + jt * 16 + colbase] = e;
        }
    }
    __syncthreads();

    // ---- Phase 2: softmax (one wave per 2 rows), write attention --------
    float* grow = attn + ((size_t)bh * S + it * 16) * S;
#pragma unroll
    for (int rr = 0; rr < 2; ++rr) {
        const int row = wave * 2 + rr;
        float* p = strip + row * LST;
        float m = -3.4e38f;
        for (int c = lane; c < S; c += 32) m = fmaxf(m, p[c]);
#pragma unroll
        for (int o = 16; o > 0; o >>= 1) m = fmaxf(m, __shfl_xor(m, o, 32));
        float sum = 0.f;
        for (int c = lane; c < S; c += 32) { float e = __expf(p[c] - m); sum += e; p[c] = e; }
#pragma unroll
        for (int o = 16; o > 0; o >>= 1) sum += __shfl_xor(sum, o, 32);
        const float inv = 1.0f / sum;
        float* g = grow + (size_t)row * S;
        for (int c = lane; c < S; c += 32) { float v = p[c] * inv; p[c] = v; g[c] = v; }
    }
    __syncthreads();

    // ---- Phase 3: x = P @ V. 8 waves = 4 n-tiles x 2 K-halves -----------
    const bf16* vb = vT + (size_t)bh * HD * S;
    const int nt    = wave & 3;
    const int kpart = wave >> 2;
    const int bcol3 = nt * 16 + colbase;        // hd
    v8f acc = {};
    const int k0 = kpart * (S / 2);
    for (int k = k0; k < k0 + (S / 2); k += 32) {
        v16bf a = load_a_lds_f32(strip, colbase, k, lane);
        v16bf b = load_b_rows(vb, S, bcol3, k, lane);
        acc = wmma_bf16(a, b, acc);
    }
    if (kpart == 1) {
#pragma unroll
        for (int r = 0; r < 8; ++r) scratch[nt * 256 + r * 32 + lane] = acc[r];
    }
    __syncthreads();
    if (kpart == 0) {
        const int d = h * HD + bcol3;
#pragma unroll
        for (int r = 0; r < 8; ++r) {
            const float v = acc[r] + scratch[nt * 256 + r * 32 + lane];
            const int s = it * 16 + rbase + r;
            xout[((size_t)(bb * S + s)) * D + d] = (bf16)v;
        }
    }
}

// ---------------------------------------------------------------------------
// Host-side launcher.
// ---------------------------------------------------------------------------
extern "C" void kernel_launch(void* const* d_in, const int* in_sizes, int n_in,
                              void* d_out, int out_size, void* d_ws, size_t ws_size,
                              hipStream_t stream) {
    const float* query = (const float*)d_in[0];
    const float* key   = (const float*)d_in[1];
    const float* value = (const float*)d_in[2];
    const int*   mask  = (const int*)  d_in[3];
    const float* Wq    = (const float*)d_in[4];
    const float* bq    = (const float*)d_in[5];
    const float* Wk    = (const float*)d_in[6];
    const float* bk    = (const float*)d_in[7];
    const float* Wv    = (const float*)d_in[8];
    const float* bv    = (const float*)d_in[9];
    const float* Wo    = (const float*)d_in[10];
    const float* bo    = (const float*)d_in[11];

    const int nBS = B * S * D;   // 8,388,608
    const int nW  = D * D;       // 1,048,576

    char* ws = (char*)d_ws;
    size_t off = 0;
    auto carve = [&](size_t elems) -> bf16* {
        bf16* p = (bf16*)(ws + off);
        off += (elems * sizeof(bf16) + 255) & ~(size_t)255;
        return p;
    };
    bf16* Xq  = carve(nBS);            // query  bf16  [B*S, D]
    bf16* Xk  = carve(nBS);
    bf16* Xv  = carve(nBS);
    bf16* Wqb = carve(nW);
    bf16* Wkb = carve(nW);
    bf16* Wvb = carve(nW);
    bf16* Wob = carve(nW);
    bf16* qbf = carve(nBS);            // q  [B,H,S,HD]
    bf16* kbf = carve(nBS);            // k  [B,H,S,HD]
    bf16* vTb = carve(nBS);            // v  [B,H,HD,S]
    bf16* xbf = carve(nBS);            // x  [B,S,D]

    // 1) fp32 -> bf16 conversions
    cvt_f32_bf16<<<(nBS + 255) / 256, 256, 0, stream>>>(query, Xq, nBS);
    cvt_f32_bf16<<<(nBS + 255) / 256, 256, 0, stream>>>(key,   Xk, nBS);
    cvt_f32_bf16<<<(nBS + 255) / 256, 256, 0, stream>>>(value, Xv, nBS);
    cvt_f32_bf16<<<(nW  + 255) / 256, 256, 0, stream>>>(Wq, Wqb, nW);
    cvt_f32_bf16<<<(nW  + 255) / 256, 256, 0, stream>>>(Wk, Wkb, nW);
    cvt_f32_bf16<<<(nW  + 255) / 256, 256, 0, stream>>>(Wv, Wvb, nW);
    cvt_f32_bf16<<<(nW  + 255) / 256, 256, 0, stream>>>(Wo, Wob, nW);

    // 2) Q/K/V projections (WMMA, 16x64 tile per wave)
    const int M = B * S, N = D, K = D;
    const int projBlocks = (M / 16) * (N / 64) / 8;   // 1024
    gemm_xwT<0><<<projBlocks, 256, 0, stream>>>(Xq, Wqb, bq, qbf, M, N, K);
    gemm_xwT<0><<<projBlocks, 256, 0, stream>>>(Xk, Wkb, bk, kbf, M, N, K);
    gemm_xwT<1><<<projBlocks, 256, 0, stream>>>(Xv, Wvb, bv, vTb, M, N, K);

    // 3) fused QK^T + mask + softmax + A@V (attention hits HBM exactly once)
    float* attn = (float*)d_out + (size_t)B * S * D;
    const size_t smemBytes = SMEM_FLOATS * sizeof(float);   // 168,192 B < 320 KB
    (void)hipFuncSetAttribute((const void*)attn_fused,
                              hipFuncAttributeMaxDynamicSharedMemorySize,
                              (int)smemBytes);
    attn_fused<<<B * H * (S / 16), 256, smemBytes, stream>>>(qbf, kbf, vTb, mask,
                                                             attn, xbf);

    // 4) out = x @ Wo^T + bo  (WMMA, fp32 out) -> d_out x region
    gemm_xwT<2><<<projBlocks, 256, 0, stream>>>(xbf, Wob, bo, d_out, M, N, K);
}
